// NormNet_12884901888474
// MI455X (gfx1250) — matrix operand order repaired and checked by reading, not compiled, not run on hardware
//
#include <hip/hip_runtime.h>
#include <math.h>

// ---------------------------------------------------------------------------
// GCN (13x GCNConv + BN + LeakyReLU) + Linear(16,3) + tanh + L2-normalize
// MI455X / gfx1250.
// Dense matmuls: f32 activations/weights pre-packed to f16 in the exact
// v_wmma_f32_16x16x32_f16 register layout (zero-padded K) -> inner loop is
// pure clause-grouped b128 loads + WMMA. Register blocking: one A fragment
// reused across NB adjacent B tiles (NB=4 for wide layers) to raise
// arithmetic intensity from 8 to ~13 FLOP/byte of fragment traffic.
// Edge aggregation: coalesced float4 gather + global_atomic_add_f32 scatter.
// Workspace (floats):
//   hbuf[N*512] | xbuf[N*512] | apack[N*256 (=N*512 halves)] |
//   bpack[131072] | dinv[N] | sum[512] | sumsq[512] | scl[512] | sft[512]
//   total ~ 515 MB
// ---------------------------------------------------------------------------

typedef __attribute__((ext_vector_type(16))) _Float16 v16h;
typedef __attribute__((ext_vector_type(8)))  float    v8f;

#define N_NODES 100000
#define N_EDGES 600000
#define BN_EPS 1e-5f
#define LRELU 0.01f

__device__ __forceinline__ void atomAddF(float* p, float v) {
  // agent-scope relaxed fp32 add -> global_atomic_add_f32 (no CAS loop)
  __hip_atomic_fetch_add(p, v, __ATOMIC_RELAXED, __HIP_MEMORY_SCOPE_AGENT);
}

// -------------------------------- utility ----------------------------------

__global__ void k_zero(float* __restrict__ p, int n) {
  int t = blockIdx.x * blockDim.x + threadIdx.x;
  if (t < n) p[t] = 0.f;
}

__global__ void k_deg_count(const int* __restrict__ dst, float* __restrict__ cnt, int E) {
  int t = blockIdx.x * blockDim.x + threadIdx.x;
  if (t < E) atomAddF(cnt + dst[t], 1.0f);
}

__global__ void k_dinv(float* __restrict__ d, int n) {
  int t = blockIdx.x * blockDim.x + threadIdx.x;
  if (t < n) {
    float deg = d[t] + 1.0f;                 // + self loop
    d[t] = (deg > 0.f) ? rsqrtf(deg) : 0.f;
  }
}

// ----------------------------- WMMA packing --------------------------------
// A fragment layout (16x32 f16, MxK):
//   lanes 0-15 : halves 0..7 -> K=+0..7,  halves 8..15 -> K=+16..23
//   lanes 16-31: halves 0..7 -> K=+8..15, halves 8..15 -> K=+24..31
// Apack index: ((mTile*kTiles + kTile)*32 + lane)*16 + j

__global__ void k_pack_a(const float* __restrict__ X, _Float16* __restrict__ Ap,
                         int N, int K, int lkt /*log2 kTiles*/) {
  int t = blockIdx.x * blockDim.x + threadIdx.x;
  int kTiles = 1 << lkt;
  int total = (N >> 4) << (9 + lkt);         // tilesM * kTiles * 512
  if (t >= total) return;
  int j    = t & 15;
  int lane = (t >> 4) & 31;
  int kt   = (t >> 9) & (kTiles - 1);
  int mT   = t >> (9 + lkt);
  int laneLo = lane & 15, hiSel = lane >> 4;
  int jj = j & 7, seg = j >> 3;
  int row = (mT << 4) + laneLo;
  int k   = (kt << 5) + seg * 16 + hiSel * 8 + jj;
  Ap[t] = (k < K) ? (_Float16)X[(size_t)row * K + k] : (_Float16)0.f;
}

// B fragment layout (32x16 f16, KxN):
//   lanes 0-15: halves 0..7 -> K=+0..7, 8..15 -> K=+16..23; lanes 16-31: K=+8..15, +24..31; lane -> column
// Bpack index: ((nTile*kTiles + kTile)*32 + lane)*16 + j

__global__ void k_pack_b(const float* __restrict__ W, _Float16* __restrict__ Bp,
                         int K, int C, int lkt) {
  int t = blockIdx.x * blockDim.x + threadIdx.x;
  int kTiles = 1 << lkt;
  int total = (C >> 4) << (9 + lkt);         // tilesN * kTiles * 512
  if (t >= total) return;
  int j    = t & 15;
  int lane = (t >> 4) & 31;
  int kt   = (t >> 9) & (kTiles - 1);
  int nT   = t >> (9 + lkt);
  int laneLo = lane & 15, hiSel = lane >> 4;
  int col = (nT << 4) + laneLo;
  int k   = (kt << 5) + ((j >> 3) << 4) + (hiSel << 3) + (j & 7);
  Bp[t] = (k < K) ? (_Float16)W[(size_t)k * C + col] : (_Float16)0.f;
}

// ------------------------------ WMMA GEMM ----------------------------------
// H[N,C] = X[N,K]*W[K,C] from packed f16 fragments, f32 accumulate.
// One wave computes a 16 x (16*NB) strip: one A fragment is reused across NB
// adjacent B tiles (register blocking). Inner loop: branch-free b128 loads +
// NB WMMAs per k-step.

template <int NB>
__global__ void k_gemm_wmma(const _Float16* __restrict__ Ap, const _Float16* __restrict__ Bp,
                            float* __restrict__ Hout, int N, int C, int kTiles) {
  const int wid  = (blockIdx.x * blockDim.x + threadIdx.x) >> 5;  // wave-uniform
  const int lane = threadIdx.x & 31;
  const int strips = (C >> 4) / NB;          // n-strips of NB tiles
  const int mT = wid / strips;
  const int nS = wid - mT * strips;
  if ((mT << 4) >= N) return;                // wave-uniform exit: EXEC full

  const _Float16* ap = Ap + (((size_t)mT * kTiles * 32) + lane) * 16;
  const _Float16* bp = Bp + (((size_t)(nS * NB) * kTiles * 32) + lane) * 16;
  const size_t bTileStride = (size_t)kTiles * 512;   // halves between n-tiles

  v8f acc[NB];
#pragma unroll
  for (int i = 0; i < NB; ++i) acc[i] = (v8f){};

#pragma unroll 2
  for (int kt = 0; kt < kTiles; ++kt) {
    v16h a = *(const v16h*)(ap + (size_t)kt * 512);
#pragma unroll
    for (int i = 0; i < NB; ++i) {
      v16h b = *(const v16h*)(bp + i * bTileStride + (size_t)kt * 512);
      acc[i] = __builtin_amdgcn_wmma_f32_16x16x32_f16(false, a, false, b,
                                                      (short)0, acc[i], false, false);
    }
  }
  // C/D layout: VGPR r -> row base+r (lanes 0-15) / base+8+r (lanes 16-31)
  const int laneLo = lane & 15, hiSel = lane >> 4;
  const int rbase = (mT << 4) + hiSel * 8;
#pragma unroll
  for (int i = 0; i < NB; ++i) {
    const int col = ((nS * NB + i) << 4) + laneLo;
#pragma unroll
    for (int r = 0; r < 8; ++r)
      Hout[(size_t)(rbase + r) * C + col] = acc[i][r];
  }
}

// --------------------------- aggregation stage -----------------------------
// out[i,:] = h[i,:]*dinv[i]^2 + bias   (self-loop message + GCN bias)

__global__ void k_seed_out(const float* __restrict__ Hm, const float* __restrict__ dinv,
                           const float* __restrict__ bias, float* __restrict__ out,
                           int N, int C, int lvC /*log2(C/4)*/) {
  int t = blockIdx.x * blockDim.x + threadIdx.x;
  if (t >= (N << lvC)) return;
  int i  = t >> lvC;
  int c4 = (t & ((1 << lvC) - 1)) << 2;
  float di = dinv[i], w = di * di;
  float4 h  = *(const float4*)(Hm + (size_t)i * C + c4);
  float4 bb = *(const float4*)(bias + c4);
  float4 o;
  o.x = h.x * w + bb.x;  o.y = h.y * w + bb.y;
  o.z = h.z * w + bb.z;  o.w = h.w * w + bb.w;
  *(float4*)(out + (size_t)i * C + c4) = o;
}

// out[dst,:] += h[src,:] * dinv[src]*dinv[dst]   (coalesced float4 gather;
// consecutive threads cover consecutive channels of one edge)

__global__ void k_edge_scatter(const float* __restrict__ Hm, const int* __restrict__ src,
                               const int* __restrict__ dst, const float* __restrict__ dinv,
                               float* __restrict__ out, int E, int C, int lvC) {
  long long t = (long long)blockIdx.x * blockDim.x + threadIdx.x;
  if (t >= ((long long)E << lvC)) return;
  int e  = (int)(t >> lvC);
  int c4 = ((int)t & ((1 << lvC) - 1)) << 2;
  int s = src[e], d = dst[e];
  float nrm = dinv[s] * dinv[d];
  float4 v = *(const float4*)(Hm + (size_t)s * C + c4);
  float* op = out + (size_t)d * C + c4;
  atomAddF(op + 0, v.x * nrm);
  atomAddF(op + 1, v.y * nrm);
  atomAddF(op + 2, v.z * nrm);
  atomAddF(op + 3, v.w * nrm);
}

// ------------------------------- BatchNorm ---------------------------------

__global__ void k_bn_stats(const float* __restrict__ Xm, float* __restrict__ sum,
                           float* __restrict__ sumsq, int N, int C, int lC, int chunks) {
  int t = blockIdx.x * blockDim.x + threadIdx.x;
  if (t >= (chunks << lC)) return;
  int c  = t & ((1 << lC) - 1);
  int ch = t >> lC;
  int rowsPer = (N + chunks - 1) / chunks;
  int r0 = ch * rowsPer;
  int r1 = r0 + rowsPer; if (r1 > N) r1 = N;
  float s = 0.f, q = 0.f;
  for (int r = r0; r < r1; ++r) {
    float v = Xm[(size_t)r * C + c];
    s += v; q += v * v;
  }
  atomAddF(sum + c, s);
  atomAddF(sumsq + c, q);
}

__global__ void k_bn_finalize(const float* __restrict__ sum, const float* __restrict__ sumsq,
                              const float* __restrict__ g, const float* __restrict__ b,
                              float* __restrict__ scl, float* __restrict__ sft,
                              int C, int N) {
  int c = blockIdx.x * blockDim.x + threadIdx.x;
  if (c >= C) return;
  float invN = 1.0f / (float)N;
  float mu  = sum[c] * invN;
  float var = sumsq[c] * invN - mu * mu;
  float is  = rsqrtf(var + BN_EPS);
  float sc  = g[c] * is;
  scl[c] = sc;
  sft[c] = b[c] - mu * sc;
}

__global__ void k_bn_apply_lrelu(float* __restrict__ Xm, const float* __restrict__ scl,
                                 const float* __restrict__ sft, int N, int C, int lvC) {
  int t = blockIdx.x * blockDim.x + threadIdx.x;
  if (t >= (N << lvC)) return;
  int i  = t >> lvC;
  int c4 = (t & ((1 << lvC) - 1)) << 2;
  float4 v = *(float4*)(Xm + (size_t)i * C + c4);
  float4 s = *(const float4*)(scl + c4);
  float4 f = *(const float4*)(sft + c4);
  float4 y;
  y.x = v.x * s.x + f.x;  y.y = v.y * s.y + f.y;
  y.z = v.z * s.z + f.z;  y.w = v.w * s.w + f.w;
  y.x = fmaxf(y.x, LRELU * y.x);  y.y = fmaxf(y.y, LRELU * y.y);
  y.z = fmaxf(y.z, LRELU * y.z);  y.w = fmaxf(y.w, LRELU * y.w);
  *(float4*)(Xm + (size_t)i * C + c4) = y;
}

// ------------------------------- final head --------------------------------
// out = normalize(tanh(x @ W16x3 + b3))

__global__ void k_head(const float* __restrict__ Xm, const float* __restrict__ Wl,
                       const float* __restrict__ bl, float* __restrict__ out, int N) {
  int i = blockIdx.x * blockDim.x + threadIdx.x;
  if (i >= N) return;
  float xv[16];
#pragma unroll
  for (int c = 0; c < 16; ++c) xv[c] = Xm[(size_t)i * 16 + c];
  float y[3];
#pragma unroll
  for (int k = 0; k < 3; ++k) {
    float acc = bl[k];
#pragma unroll
    for (int c = 0; c < 16; ++c) acc += xv[c] * Wl[c * 3 + k];
    y[k] = tanhf(acc);
  }
  float nrm = sqrtf(y[0] * y[0] + y[1] * y[1] + y[2] * y[2]) + 1e-12f;
#pragma unroll
  for (int k = 0; k < 3; ++k) out[(size_t)i * 3 + k] = y[k] / nrm;
}

// ------------------------------- launcher ----------------------------------

extern "C" void kernel_launch(void* const* d_in, const int* in_sizes, int n_in,
                              void* d_out, int out_size, void* d_ws, size_t ws_size,
                              hipStream_t stream) {
  (void)in_sizes; (void)n_in; (void)out_size; (void)ws_size;
  static const int Hd[15] = {7, 16, 32, 64, 128, 256, 256, 512, 256, 256, 128, 64, 32, 16, 3};
  const int N = N_NODES, E = N_EDGES;

  const float* x0  = (const float*)d_in[0];
  const int*   ei  = (const int*)d_in[1];
  const int*   src = ei;
  const int*   dst = ei + E;
  const float* gw[13]; const float* gb[13]; const float* bg[13]; const float* bbp[13];
  for (int i = 0; i < 13; ++i) {
    gw[i]  = (const float*)d_in[2 + i];
    gb[i]  = (const float*)d_in[15 + i];
    bg[i]  = (const float*)d_in[28 + i];
    bbp[i] = (const float*)d_in[41 + i];
  }
  const float* lw = (const float*)d_in[54];
  const float* lb = (const float*)d_in[55];

  float* ws = (float*)d_ws;
  const size_t BUF = (size_t)N * 512;
  float*     hbuf  = ws;
  float*     xbuf  = ws + BUF;
  _Float16*  apack = (_Float16*)(ws + 2 * BUF);            // N*512 halves
  _Float16*  bpack = (_Float16*)(ws + 2 * BUF + BUF / 2);  // 512*512 halves
  float*     dinv  = ws + 2 * BUF + BUF / 2 + (512 * 512) / 2;
  float*     sum   = dinv + N;
  float*     sumsq = sum + 512;
  float*     scl   = sumsq + 512;
  float*     sft   = scl + 512;

  // ---- symmetric normalization coefficients ----
  k_zero<<<(N + 255) / 256, 256, 0, stream>>>(dinv, N);
  k_deg_count<<<(E + 255) / 256, 256, 0, stream>>>(dst, dinv, E);
  k_dinv<<<(N + 255) / 256, 256, 0, stream>>>(dinv, N);

  const float* X = x0;
  for (int l = 0; l < 13; ++l) {
    int K = Hd[l], C = Hd[l + 1];
    int Kp = (K + 31) & ~31;                 // pad K to multiple of 32
    int kTiles = Kp >> 5;
    int lkt = __builtin_ctz(kTiles);
    int lC  = __builtin_ctz(C);
    int lvC = lC - 2;                        // log2(C/4)

    // pack A (activations) and B (weights) into WMMA fragment layout, f16
    int aTot = (N >> 4) << (9 + lkt);
    int bTot = (C >> 4) << (9 + lkt);
    k_pack_a<<<(aTot + 255) / 256, 256, 0, stream>>>(X, apack, N, K, lkt);
    k_pack_b<<<(bTot + 255) / 256, 256, 0, stream>>>(gw[l], bpack, K, C, lkt);

    // h = X @ W   (WMMA; one A fragment reused across NB B tiles per wave)
    int tilesN = C / 16;
    int NB = (tilesN >= 4) ? 4 : tilesN;     // 4 for C>=64, 2 for C=32, 1 for C=16
    int waves = (N / 16) * (tilesN / NB);
    int thr = waves * 32;
    int blocks = (thr + 255) / 256;
    if (NB == 4)      k_gemm_wmma<4><<<blocks, 256, 0, stream>>>(apack, bpack, hbuf, N, C, kTiles);
    else if (NB == 2) k_gemm_wmma<2><<<blocks, 256, 0, stream>>>(apack, bpack, hbuf, N, C, kTiles);
    else              k_gemm_wmma<1><<<blocks, 256, 0, stream>>>(apack, bpack, hbuf, N, C, kTiles);

    // out = self-loop msg + bias, then scatter edge messages
    int tot4 = N << lvC;
    k_seed_out<<<(tot4 + 255) / 256, 256, 0, stream>>>(hbuf, dinv, gb[l], xbuf, N, C, lvC);
    long long et = (long long)E << lvC;
    k_edge_scatter<<<(unsigned)((et + 255) / 256), 256, 0, stream>>>(
        hbuf, src, dst, dinv, xbuf, E, C, lvC);

    // BatchNorm (training-mode batch stats) + LeakyReLU, in place
    k_zero<<<4, 256, 0, stream>>>(sum, 1024);  // sum+sumsq contiguous
    const int chunks = 256;
    int st = chunks << lC;
    k_bn_stats<<<(st + 255) / 256, 256, 0, stream>>>(xbuf, sum, sumsq, N, C, lC, chunks);
    k_bn_finalize<<<(C + 63) / 64, 64, 0, stream>>>(sum, sumsq, bg[l], bbp[l], scl, sft, C, N);
    k_bn_apply_lrelu<<<(tot4 + 255) / 256, 256, 0, stream>>>(xbuf, scl, sft, N, C, lvC);

    X = xbuf;  // xbuf is the layer output; hbuf/apack free for next layer
  }

  k_head<<<(N + 255) / 256, 256, 0, stream>>>(xbuf, lw, lb, (float*)d_out, N);
}
